// rhs_base_63531156242763
// MI455X (gfx1250) — compile-verified, matrix operand order checked by reading, not run on hardware
//
#include <hip/hip_runtime.h>
#include <hip/hip_bf16.h>

#define NE 8
#define DIM 1024
#define DHID 4096
#define NB 4096
#define TM 16              // samples per tile
#define HC 512             // hidden mega-chunk
#define YS (DIM + 8)       // y LDS row stride (bf16) -> conflict-free
#define HS (HC + 8)        // hid LDS row stride
#define KB1 (DIM / 32)     // 32  k-blocks in GEMM1
#define KB2 (DHID / 32)    // 128 k-blocks in GEMM2

typedef __attribute__((ext_vector_type(16))) __bf16 v16bf;
typedef __attribute__((ext_vector_type(8)))  __bf16 v8bf;
typedef __attribute__((ext_vector_type(8)))  float  v8f;

__device__ __forceinline__ __bf16 f2bf(float x) { return (__bf16)x; }

__device__ __forceinline__ v16bf joinbf(v8bf lo, v8bf hi) {
  return __builtin_shufflevector(lo, hi, 0,1,2,3,4,5,6,7,8,9,10,11,12,13,14,15);
}

__device__ __forceinline__ float fast_tanh(float x) {
  x = fminf(15.0f, fmaxf(-15.0f, x));
  float e = __expf(-2.0f * x);
  return (1.0f - e) * __builtin_amdgcn_rcpf(1.0f + e);
}

__device__ __forceinline__ float sigmoidf_(float x) {
  return __builtin_amdgcn_rcpf(1.0f + __expf(-x));
}

// ---------------- bucket kernels ----------------
__global__ void bucket_zero(int* counts) {
  if (threadIdx.x < NE) counts[threadIdx.x] = 0;
}

__global__ void bucket_build(const float* __restrict__ t, int* counts,
                             int* __restrict__ lists) {
  int b = blockIdx.x * blockDim.x + threadIdx.x;
  if (b >= NB) return;
  int e = (int)(t[b] * (float)NE);                 // t / H_STEP, H_STEP = 1/8
  e = e < 0 ? 0 : (e > NE - 1 ? NE - 1 : e);
  int pos = atomicAdd(&counts[e], 1);
  lists[e * NB + pos] = b;
}

// ---------------- weight pre-convert: f32 row-major -> bf16 WMMA-B tiles ----
// Tile (ntile,kb) holds B-fragment for cols [ntile*16,+16), rows [kb*32,+32):
// 1 KB per tile; lane L owns 32 B: col = ntile*16+(L&15), kHi = (L>=16)*8,
// elems 0..7 = rows k0+kHi+0..7, elems 8..15 = rows k0+kHi+16..23.
__global__ __launch_bounds__(256) void convert_tiles(
    const float* __restrict__ src, __bf16* __restrict__ dst,
    int K, int N, int KB)
{
  size_t gid = (size_t)blockIdx.x * blockDim.x + threadIdx.x;
  size_t perExpert = (size_t)K * N / 16;           // 16 elems per thread
  int e = (int)(gid / perExpert);
  size_t r = gid - (size_t)e * perExpert;
  int lane = (int)(r & 31);
  size_t tile = r >> 5;
  int ntile = (int)(tile / (size_t)KB);
  int kb = (int)(tile - (size_t)ntile * KB);
  int col = ntile * 16 + (lane & 15);
  int k0 = kb * 32 + ((lane >> 4) << 3);
  const float* in = src + (size_t)e * K * N;
  v8bf lo, hi;
  #pragma unroll
  for (int i = 0; i < 8; ++i) {
    lo[i] = f2bf(in[(size_t)(k0 + i) * N + col]);
    hi[i] = f2bf(in[(size_t)(k0 + 16 + i) * N + col]);
  }
  v8bf* o = (v8bf*)(dst + (size_t)e * K * N + (tile << 9) + (lane << 4));
  o[0] = lo;
  o[1] = hi;
}

// ---------------- fused per-expert MLP ----------------
__global__ __launch_bounds__(256) void moe_kernel(
    const float* __restrict__ y,
    const __bf16* __restrict__ W1t, const float* __restrict__ b1,
    const __bf16* __restrict__ W2t, const float* __restrict__ b2,
    const float* __restrict__ scales, const float* __restrict__ shifta,
    const float* __restrict__ shiftb, const int* __restrict__ counts,
    const int* __restrict__ lists, float* __restrict__ out)
{
  __shared__ __bf16 yls[TM * YS];
  __shared__ __bf16 hls[TM * HS];
  __shared__ int rows[TM];

  const int e = blockIdx.y;
  const int cnt = counts[e];
  const int tileBase = blockIdx.x * TM;
  if (tileBase >= cnt) return;

  const int tid  = threadIdx.x;
  const int wave = tid >> 5;
  const int lane = tid & 31;
  const int nlo  = lane & 15;           // A row / C,B column within tile
  const int kHi  = (lane >> 4) << 3;    // per-lane-half K offset
  const int mHi  = (lane >> 4) << 3;    // per-lane-half C row offset

  if (tid < TM) {
    int gi = tileBase + tid;
    rows[tid] = (gi < cnt) ? lists[e * NB + gi] : -1;
  }
  __syncthreads();

  // stage y tile to LDS as bf16 (zero-pad invalid rows)
  for (int i = tid; i < TM * DIM; i += 256) {
    int m = i >> 10;
    int k = i & (DIM - 1);
    int r = rows[m];
    yls[m * YS + k] = (r >= 0) ? f2bf(y[(size_t)r * DIM + k]) : f2bf(0.0f);
  }
  __syncthreads();

  const __bf16* W1e = W1t + (size_t)e * DIM * DHID;   // tiled bf16
  const __bf16* W2e = W2t + (size_t)e * DHID * DIM;   // tiled bf16

  // output accumulators: wave owns D columns [wave*128, +128)
  v8f acc2[8];
  #pragma unroll
  for (int j = 0; j < 8; ++j)
    #pragma unroll
    for (int r = 0; r < 8; ++r) acc2[j][r] = 0.0f;

  for (int hc = 0; hc < DHID / HC; ++hc) {
    // ---- phase 1: hid chunk [16 x HC]; wave computes 64 cols (4 tiles) ----
    v8f acc1[4];
    #pragma unroll
    for (int nt = 0; nt < 4; ++nt) {
      const float bias = b1[(size_t)e * DHID + hc * HC + wave * 64 + nt * 16 + nlo];
      #pragma unroll
      for (int r = 0; r < 8; ++r) acc1[nt][r] = bias;
    }
    const int hTile0 = (hc * HC + wave * 64) >> 4;     // first h-tile this wave
    for (int k = 0; k < DIM; k += 32) {
      const __bf16* pa = &yls[nlo * YS + k + kHi];
      v16bf a = joinbf(*(const v8bf*)pa, *(const v8bf*)(pa + 16));
      #pragma unroll
      for (int nt = 0; nt < 4; ++nt) {
        const __bf16* pb = W1e + ((((size_t)(hTile0 + nt)) * KB1 + (k >> 5)) << 9)
                               + (lane << 4);
        v16bf bfr = joinbf(*(const v8bf*)pb, *(const v8bf*)(pb + 8));
        acc1[nt] = __builtin_amdgcn_wmma_f32_16x16x32_bf16(
            false, a, false, bfr, (short)0, acc1[nt], false, false);
      }
    }
    // tanh + store bf16 hid tiles to LDS
    #pragma unroll
    for (int nt = 0; nt < 4; ++nt) {
      const int hLocal = wave * 64 + nt * 16 + nlo;
      #pragma unroll
      for (int r = 0; r < 8; ++r)
        hls[(r + mHi) * HS + hLocal] = f2bf(fast_tanh(acc1[nt][r]));
    }
    __syncthreads();

    // ---- phase 2: out += hid_chunk @ W2[chunk rows, :] ----
    for (int k2 = 0; k2 < HC; k2 += 32) {
      const __bf16* pa = &hls[nlo * HS + k2 + kHi];
      v16bf a = joinbf(*(const v8bf*)pa, *(const v8bf*)(pa + 16));
      const int kbG = (hc * HC + k2) >> 5;
      __builtin_prefetch(W2e + ((((size_t)(wave * 8)) * KB2 + kbG + 1) << 9), 0, 1);
      #pragma unroll
      for (int j = 0; j < 8; ++j) {
        const __bf16* pb = W2e + ((((size_t)(wave * 8 + j)) * KB2 + kbG) << 9)
                               + (lane << 4);
        v16bf bfr = joinbf(*(const v8bf*)pb, *(const v8bf*)(pb + 8));
        acc2[j] = __builtin_amdgcn_wmma_f32_16x16x32_bf16(
            false, a, false, bfr, (short)0, acc2[j], false, false);
      }
    }
    __syncthreads();
  }

  // ---- epilogue: b2, sigmoid(scales), eigen-affine with y ----
  const float av = -sigmoidf_(shifta[0]);   // EIGINIT + sig*(EIGMIN-EIGINIT)
  const float bv =  sigmoidf_(shiftb[0]);   // EIGINIT + sig*(EIGMAX-EIGINIT)
  const float c0 = 0.5f * (bv - av);
  const float c1 = 0.5f * (av + bv);

  #pragma unroll
  for (int j = 0; j < 8; ++j) {
    const int dCol = wave * 128 + j * 16 + nlo;
    const float ss  = sigmoidf_(scales[dCol]);
    const float bb2 = b2[(size_t)e * DIM + dCol];
    #pragma unroll
    for (int r = 0; r < 8; ++r) {
      int gb = rows[r + mHi];
      if (gb >= 0) {
        float f  = acc2[j][r] + bb2;
        float yv = y[(size_t)gb * DIM + dCol];
        out[(size_t)gb * DIM + dCol] = c0 * ss * f + c1 * yv;
      }
    }
  }
}

extern "C" void kernel_launch(void* const* d_in, const int* in_sizes, int n_in,
                              void* d_out, int out_size, void* d_ws, size_t ws_size,
                              hipStream_t stream) {
  const float* t      = (const float*)d_in[0];
  const float* y      = (const float*)d_in[1];
  const float* W1     = (const float*)d_in[2];
  const float* b1     = (const float*)d_in[3];
  const float* W2     = (const float*)d_in[4];
  const float* b2     = (const float*)d_in[5];
  const float* scales = (const float*)d_in[6];
  const float* shifta = (const float*)d_in[7];
  const float* shiftb = (const float*)d_in[8];
  float* out = (float*)d_out;

  // workspace layout
  const size_t wBytes = (size_t)NE * DIM * DHID * sizeof(__bf16);  // 64 MB each
  int*    counts = (int*)d_ws;                                     // 8 ints
  int*    lists  = (int*)((char*)d_ws + 1024);                     // NE*NB ints
  __bf16* W1t    = (__bf16*)((char*)d_ws + (1u << 18));            // 256 KB in
  __bf16* W2t    = (__bf16*)((char*)d_ws + (1u << 18) + wBytes);

  // weight pre-conversion (streaming, bf16 fragment-tiled)
  const int cBlocks = (int)((size_t)NE * DIM * DHID / 16 / 256);   // 8192
  convert_tiles<<<cBlocks, 256, 0, stream>>>(W1, W1t, DIM, DHID, KB1);
  convert_tiles<<<cBlocks, 256, 0, stream>>>(W2, W2t, DHID, DIM, KB2);

  bucket_zero<<<1, 64, 0, stream>>>(counts);
  bucket_build<<<NB / 256, 256, 0, stream>>>(t, counts, lists);

  dim3 grid(NB / TM, NE);
  moe_kernel<<<grid, 256, 0, stream>>>(y, W1t, b1, W2t, b2, scales,
                                       shifta, shiftb, counts, lists, out);
}
